// AttentionBasedPruner_19078244729170
// MI455X (gfx1250) — matrix operand order repaired and checked by reading, not compiled, not run on hardware
//
#include <hip/hip_runtime.h>

typedef __attribute__((ext_vector_type(16))) _Float16 v16h;
typedef __attribute__((ext_vector_type(8)))  float    v8f;

#define BATCH     32
#define NTOK      4096
#define DIM       768
#define HDIM      192
#define NKEEP     2867          // int(4096 * 0.7)
#define KIDX      (NTOK - NKEEP) // 1229: threshold element in ascending sort
#define KT_CNT    24            // 768 / 32
#define HT_HALF   6             // 6 h-tiles of 16 per workgroup half (96 ch)
#define TILE_HALF (HT_HALF * KT_CNT * 32 * 16)   // 73728 halfs per (hi|lo) plane
#define W1P_HALF  (2 * TILE_HALF)                // per h-half: hi plane + lo plane
#define GROUPS    ((BATCH * NTOK) / 16)          // 8192 token-tiles

// ---------------------------------------------------------------------------
// Pack w1 [768][192] f32 into WMMA A-operand lane layout, split hi/lo f16.
// Layout: [hh(2)][part(2: hi,lo)][ht(6)][kt(24)][lane(32)][e(16)]
// A is w1^T tile (M=h 16, K=d 32). 16-bit A 16x32 lane layout:
//   lane=(hi<<4)|l : M=l ; e -> v=e>>1 ; base = v<4 ? 2v : 16+2(v-4)
//   K = base + (e&1) + (hi?8:0)
// ---------------------------------------------------------------------------
__global__ void pack_w1_kernel(const float* __restrict__ w1,
                               _Float16* __restrict__ w1p) {
  int t = blockIdx.x * blockDim.x + threadIdx.x;
  if (t >= 2 * W1P_HALF) return;
  int e    = t & 15;
  int lane = (t >> 4) & 31;
  int kt   = (t >> 9) % KT_CNT;
  int r1   = (t >> 9) / KT_CNT;
  int ht   = r1 % HT_HALF;
  int r2   = r1 / HT_HALF;
  int part = r2 & 1;
  int hh   = r2 >> 1;
  int l  = lane & 15;
  int hb = lane >> 4;
  int v  = e >> 1;
  int base = (v < 4) ? (2 * v) : (16 + 2 * (v - 4));
  int K  = base + (e & 1) + (hb ? 8 : 0);
  int d  = kt * 32 + K;
  int h  = hh * 96 + ht * 16 + l;
  float    val = w1[d * HDIM + h];
  _Float16 hi  = (_Float16)val;
  w1p[t] = (part == 0) ? hi : (_Float16)(val - (float)hi);
}

// ---------------------------------------------------------------------------
// Fused score GEMM: partial[hh][b][tok] = sum_{h in half} gelu(z)*w2[h]
// grid.x = 512 (2 h-halves x 256 token slots), 256 threads (8 waves),
// dynamic LDS = 589824 B requested; w1 hi+lo planes for this h-half use 294912 B.
// LDS fill uses gfx1250 async global->LDS DMA (ASYNCcnt), no VGPR roundtrip.
// ---------------------------------------------------------------------------
__global__ void score_kernel(const float* __restrict__ x,
                             const _Float16* __restrict__ w1p,
                             const float* __restrict__ b1,
                             const float* __restrict__ w2,
                             float* __restrict__ part) {
  extern __shared__ char smem_raw[];
  _Float16* lds = (_Float16*)smem_raw;   // [part][ht][kt][lane][16]

  const int hh = blockIdx.x & 1;

  // async copy of this half's packed w1 (288 KB) into LDS:
  // global_load_async_to_lds_b128 <lds_byte_addr>, <voffset>, <sbase>
  {
    const _Float16* gsrc = w1p + (size_t)hh * W1P_HALF;
    const unsigned ldsbase = (unsigned)(unsigned long long)(void*)lds;
    const int n16 = (W1P_HALF * 2) / 16;          // 18432 x 16B transfers
    for (int i = threadIdx.x; i < n16; i += blockDim.x) {
      unsigned loff = ldsbase + (unsigned)i * 16u;
      unsigned goff = (unsigned)i * 16u;
      asm volatile("global_load_async_to_lds_b128 %0, %1, %2"
                   :: "v"(loff), "v"(goff), "s"(gsrc) : "memory");
    }
    asm volatile("s_wait_asynccnt 0" ::: "memory");
  }
  __syncthreads();

  const int lane = threadIdx.x & 31;
  const int n    = lane & 15;     // token column
  const int hb   = lane >> 4;     // K-half / M-half selector
  const int slot = blockIdx.x >> 1;
  const int wid  = slot * 8 + (threadIdx.x >> 5);
  const int wstr = (gridDim.x >> 1) * 8;

  for (int g = wid; g < GROUPS; g += wstr) {
    const int b    = g >> 8;
    const int tok0 = (g & 255) << 4;
    const int tok  = tok0 + n;
    const float* xrow = x + (size_t)(b * NTOK + tok) * DIM + hb * 8;

    v8f acc[HT_HALF] = {};

    for (int kt = 0; kt < KT_CNT; ++kt) {
      // B operand: lane holds x[tok][kt*32 + hb*8 .. +8) and [kt*32 + 16 + hb*8 .. +8), split into hi/lo f16
      const float4* xp = (const float4*)(xrow + kt * 32);
      float4 q0 = xp[0], q1 = xp[1], q2 = xp[4], q3 = xp[5];
      v16h bh, bl;
#define CVT(i, VAL) { float _v = (VAL); _Float16 _h = (_Float16)_v; \
                      bh[i] = _h; bl[i] = (_Float16)(_v - (float)_h); }
      CVT(0,  q0.x) CVT(1,  q0.y) CVT(2,  q0.z) CVT(3,  q0.w)
      CVT(4,  q1.x) CVT(5,  q1.y) CVT(6,  q1.z) CVT(7,  q1.w)
      CVT(8,  q2.x) CVT(9,  q2.y) CVT(10, q2.z) CVT(11, q2.w)
      CVT(12, q3.x) CVT(13, q3.y) CVT(14, q3.z) CVT(15, q3.w)
#undef CVT

#pragma unroll
      for (int ht = 0; ht < HT_HALF; ++ht) {
        const int toff = ((ht * KT_CNT + kt) * 32 + lane) * 16;
        v16h ah = *(const v16h*)(lds + toff);
        v16h al = *(const v16h*)(lds + TILE_HALF + toff);
        // split-f16: (ah+al)(bh+bl) ~= al*bh + ah*bl + ah*bh  (f32 accum)
        acc[ht] = __builtin_amdgcn_wmma_f32_16x16x32_f16(
            false, al, false, bh, (short)0, acc[ht], false, false);
        acc[ht] = __builtin_amdgcn_wmma_f32_16x16x32_f16(
            false, ah, false, bl, (short)0, acc[ht], false, false);
        acc[ht] = __builtin_amdgcn_wmma_f32_16x16x32_f16(
            false, ah, false, bh, (short)0, acc[ht], false, false);
      }
    }

    // epilogue: D[m][n] lane layout -> lane(n,hb) VGPR r = z[tok n][h = ht*16+hb*8+r]
    float partial = 0.0f;
#pragma unroll
    for (int ht = 0; ht < HT_HALF; ++ht) {
#pragma unroll
      for (int r = 0; r < 8; ++r) {
        const int h = hh * 96 + ht * 16 + hb * 8 + r;
        float z  = acc[ht][r] + b1[h];
        float ge = 0.5f * z * (1.0f + erff(z * 0.70710678118654752f));
        partial += ge * w2[h];
      }
    }
    partial += __shfl_xor(partial, 16, 32);
    if (hb == 0)
      part[(size_t)hh * (BATCH * NTOK) + b * NTOK + tok0 + n] = partial;
  }
}

// ---------------------------------------------------------------------------
// Per-batch top-k via bitonic sort (threshold) + prefix scan (ascending idx).
// grid.x = 32, 1024 threads.
// ---------------------------------------------------------------------------
__global__ void topk_kernel(const float* __restrict__ part,
                            const float* __restrict__ b2,
                            int* __restrict__ idx) {
  __shared__ float vals[NTOK];
  __shared__ float sv[NTOK];
  __shared__ int   psum[1024];

  const int b   = blockIdx.x;
  const int tid = threadIdx.x;
  const float* p0 = part + b * NTOK;
  const float* p1 = part + BATCH * NTOK + b * NTOK;
  const float b2v = b2[0];

  for (int i = tid; i < NTOK; i += 1024) {
    float v = p0[i] + p1[i] + b2v;
    vals[i] = v;
    sv[i]   = v;
  }
  __syncthreads();

  // bitonic sort ascending (4096 elems, 1024 threads)
  for (int k = 2; k <= NTOK; k <<= 1) {
    for (int j = k >> 1; j > 0; j >>= 1) {
      for (int i = tid; i < NTOK; i += 1024) {
        int p = i ^ j;
        if (p > i) {
          float a = sv[i], c = sv[p];
          bool up = ((i & k) == 0);
          if ((a > c) == up) { sv[i] = c; sv[p] = a; }
        }
      }
      __syncthreads();
    }
  }
  const float T = sv[KIDX];   // 2867th-largest score

  // per-thread counts over 4 consecutive elements
  const int i0 = tid * 4;
  float v4[4];
  int lgt = 0, leq = 0;
#pragma unroll
  for (int q = 0; q < 4; ++q) {
    float v = vals[i0 + q];
    v4[q] = v;
    lgt += (v > T);
    leq += (v == T);
  }
  psum[tid] = (lgt << 13) | leq;   // both counts <= 4096 -> 13 bits each
  __syncthreads();

  // Hillis-Steele inclusive scan over 1024 packed sums
  for (int off = 1; off < 1024; off <<= 1) {
    int vcur = psum[tid];
    int vadd = (tid >= off) ? psum[tid - off] : 0;
    __syncthreads();
    psum[tid] = vcur + vadd;
    __syncthreads();
  }
  const int total    = psum[1023];
  const int count_gt = total >> 13;
  const int need_eq  = NKEEP - count_gt;   // ties kept by lowest index

  int gt_before = (psum[tid] >> 13) - lgt;
  int eq_before = (psum[tid] & 8191) - leq;
#pragma unroll
  for (int q = 0; q < 4; ++q) {
    float v = v4[q];
    if (v > T) {
      int eqk = (eq_before < need_eq) ? eq_before : need_eq;
      idx[b * NKEEP + gt_before + eqk] = i0 + q;
      ++gt_before;
    } else if (v == T) {
      if (eq_before < need_eq)
        idx[b * NKEEP + gt_before + eq_before] = i0 + q;
      ++eq_before;
    }
  }
}

// ---------------------------------------------------------------------------
// Gather kept rows: one block per output row, float4 copies.
// ---------------------------------------------------------------------------
__global__ void gather_kernel(const float* __restrict__ x,
                              const int* __restrict__ idx,
                              float* __restrict__ out) {
  const int row = blockIdx.x;
  const int b   = row / NKEEP;
  const int tokIdx = idx[row];
  const float4* src = (const float4*)(x + (size_t)(b * NTOK + tokIdx) * DIM);
  float4*       dst = (float4*)(out + (size_t)row * DIM);
  dst[threadIdx.x] = src[threadIdx.x];
}

// ---------------------------------------------------------------------------
extern "C" void kernel_launch(void* const* d_in, const int* in_sizes, int n_in,
                              void* d_out, int out_size, void* d_ws, size_t ws_size,
                              hipStream_t stream) {
  const float* x  = (const float*)d_in[0];
  const float* w1 = (const float*)d_in[1];
  const float* b1 = (const float*)d_in[2];
  const float* w2 = (const float*)d_in[3];
  const float* b2 = (const float*)d_in[4];
  float* out = (float*)d_out;

  char* ws = (char*)d_ws;
  _Float16* w1p   = (_Float16*)ws;                        // 2*294912 halfs = 589824 B
  float*    partb = (float*)(ws + 589824);                // 2*131072 f32  = 1048576 B
  int*      idx   = (int*)(ws + 589824 + 1048576);        // 91744 ints    = 366976 B

  // 1) pack + split w1 into WMMA A-operand lane layout
  pack_w1_kernel<<<(2 * W1P_HALF + 255) / 256, 256, 0, stream>>>(w1, w1p);

  // 2) fused split-f16 WMMA GEMM + GELU + w2 reduction -> partial scores
  score_kernel<<<512, 256, 2 * W1P_HALF * (int)sizeof(_Float16), stream>>>(
      x, w1p, b1, w2, partb);

  // 3) per-batch top-k threshold + ascending index emission
  topk_kernel<<<BATCH, 1024, 0, stream>>>(partb, b2, idx);

  // 4) gather kept rows
  gather_kernel<<<BATCH * NKEEP, DIM / 4, 0, stream>>>(x, idx, out);
}
